// ProfessionalLlama4Attention_87617332839001
// MI455X (gfx1250) — matrix operand order repaired and checked by reading, not compiled, not run on hardware
//
#include <hip/hip_runtime.h>
#include <hip/hip_bf16.h>
#include <stdint.h>

// ---------------------------------------------------------------------------
// Types for CDNA5 WMMA (wave32): v_wmma_f32_16x16x32_bf16
// ---------------------------------------------------------------------------
typedef __attribute__((ext_vector_type(16))) __bf16 v16bf;
typedef __attribute__((ext_vector_type(8)))  float  v8f;
typedef __attribute__((ext_vector_type(4)))  int    v4i;

union ABFrag {
    v16bf v;
    uint4 q[2];
};

__device__ __forceinline__ v8f wmma_bf16(const ABFrag& a, const ABFrag& b, v8f c) {
    return __builtin_amdgcn_wmma_f32_16x16x32_bf16(
        /*neg_a=*/false, a.v, /*neg_b=*/false, b.v,
        /*c_mod=*/(short)0, c, /*reuse_a=*/false, /*reuse_b=*/false);
}

__device__ __forceinline__ unsigned short f32_to_bf16_rne(float f) {
    unsigned int u = __float_as_uint(f);
    u += 0x7FFFu + ((u >> 16) & 1u);
    return (unsigned short)(u >> 16);
}

// A-fragment (16x32 bf16): lane holds row = lane%16; VGPR0-3: K = 8*(lane/16)..+7,
// VGPR4-7: K = 16 + 8*(lane/16)..+7.  p must point at (row, kchunk + 8*(lane/16)).
__device__ __forceinline__ ABFrag load_a_frag(const unsigned short* p) {
    ABFrag f;
    f.q[0] = *reinterpret_cast<const uint4*>(p);
    f.q[1] = *reinterpret_cast<const uint4*>(p + 16);
    return f;
}

// B-fragment (32x16 bf16): lane holds col = lane%16; VGPR0-3: K = 8*(lane/16)..+7,
// VGPR4-7: K = 16 + 8*(lane/16)..+7.  p must point at (col, kchunk + 8*(lane/16)).
__device__ __forceinline__ ABFrag load_b_frag(const unsigned short* p) {
    ABFrag f;
    f.q[0] = *reinterpret_cast<const uint4*>(p);
    f.q[1] = *reinterpret_cast<const uint4*>(p + 16);
    return f;
}

// ---------------------------------------------------------------------------
// Async global->LDS copy (gfx1250 ASYNCcnt path), with portable fallback.
// Builtin signature (probed via round-2 diagnostic): takes non-const typed
// pointers: (v4i addrspace(1)*, v4i addrspace(3)*, imm offset, imm cpol).
// ---------------------------------------------------------------------------
#if defined(__has_builtin)
#if __has_builtin(__builtin_amdgcn_global_load_async_to_lds_b128)
#define HAS_ASYNC_LDS 1
#endif
#endif

#define GLOBAL_AS __attribute__((address_space(1)))
#define LDS_AS    __attribute__((address_space(3)))

__device__ __forceinline__ void async_cp16(const unsigned short* g, unsigned short* l) {
#if defined(HAS_ASYNC_LDS)
    // generic->AS casts via uintptr_t (LDS generic addr low 32 bits == LDS offset)
    __builtin_amdgcn_global_load_async_to_lds_b128(
        (GLOBAL_AS v4i*)(uintptr_t)g,
        (LDS_AS v4i*)(uintptr_t)l,
        0, 0);
#else
    *reinterpret_cast<uint4*>(l) = *reinterpret_cast<const uint4*>(g);
#endif
}

#if defined(HAS_ASYNC_LDS)
#define WAIT_ASYNC(n) asm volatile("s_wait_asynccnt %0" ::"i"(n) : "memory")
#else
#define WAIT_ASYNC(n)
#endif

// ---------------------------------------------------------------------------
// Constants for this problem instance
// ---------------------------------------------------------------------------
#define SEQ      2048
#define HID      4096
#define NQH      32
#define NKVH     8
#define HD       128
#define KVDIM    (NKVH * HD)   // 1024

// ---------------------------------------------------------------------------
// 1) fp32 -> bf16 convert (vectorized x4)
// ---------------------------------------------------------------------------
__global__ __launch_bounds__(256) void cvt_f32_bf16(const float* __restrict__ in,
                                                    unsigned short* __restrict__ out,
                                                    int n4) {
    int i = blockIdx.x * 256 + threadIdx.x;
    if (i < n4) {
        float4 v = reinterpret_cast<const float4*>(in)[i];
        ushort4 r;
        r.x = f32_to_bf16_rne(v.x);
        r.y = f32_to_bf16_rne(v.y);
        r.z = f32_to_bf16_rne(v.z);
        r.w = f32_to_bf16_rne(v.w);
        reinterpret_cast<ushort4*>(out)[i] = r;
    }
}

// ---------------------------------------------------------------------------
// 2) GEMM:  C[m,n] = sum_k A[m,k] * B[n,k]   (A,B bf16 row-major, C fp32)
//    Each wave: 32(M) x 64(N) tile, K-loop step 32, 8 WMMAs per step.
//    Relies on the 192MB L2 for B reuse (whole bf16 weight set is L2-resident).
// ---------------------------------------------------------------------------
__global__ __launch_bounds__(256) void gemm_nt_bf16_f32(const unsigned short* __restrict__ A,
                                                        const unsigned short* __restrict__ B,
                                                        float* __restrict__ C,
                                                        int M, int N, int K) {
    const int lane = threadIdx.x & 31;
    const int wave = threadIdx.x >> 5;
    const int half = lane >> 4;
    const int ln   = lane & 15;

    const int tilesN = N >> 6;
    int id = blockIdx.x * 8 + wave;
    int tm = id / tilesN;
    int tn = id - tm * tilesN;
    const int row0 = tm * 32;
    const int col0 = tn * 64;

    v8f acc[2][4];
    for (int i = 0; i < 2; ++i)
        for (int j = 0; j < 4; ++j)
            for (int r = 0; r < 8; ++r) acc[i][j][r] = 0.f;

    const unsigned short* aBase[2];
    aBase[0] = A + (size_t)(row0 + ln) * K + 8 * half;
    aBase[1] = A + (size_t)(row0 + 16 + ln) * K + 8 * half;
    const unsigned short* bBase[4];
    for (int j = 0; j < 4; ++j)
        bBase[j] = B + (size_t)(col0 + 16 * j + ln) * K + 8 * half;

    for (int kk = 0; kk < K; kk += 32) {
        ABFrag a0 = load_a_frag(aBase[0] + kk);
        ABFrag a1 = load_a_frag(aBase[1] + kk);
        ABFrag b0 = load_b_frag(bBase[0] + kk);
        ABFrag b1 = load_b_frag(bBase[1] + kk);
        ABFrag b2 = load_b_frag(bBase[2] + kk);
        ABFrag b3 = load_b_frag(bBase[3] + kk);
        acc[0][0] = wmma_bf16(a0, b0, acc[0][0]);
        acc[0][1] = wmma_bf16(a0, b1, acc[0][1]);
        acc[0][2] = wmma_bf16(a0, b2, acc[0][2]);
        acc[0][3] = wmma_bf16(a0, b3, acc[0][3]);
        acc[1][0] = wmma_bf16(a1, b0, acc[1][0]);
        acc[1][1] = wmma_bf16(a1, b1, acc[1][1]);
        acc[1][2] = wmma_bf16(a1, b2, acc[1][2]);
        acc[1][3] = wmma_bf16(a1, b3, acc[1][3]);
    }

    // C/D layout: element (Mr, Nc): lane = Nc + 16*(Mr/8), VGPR = Mr%8
    for (int i = 0; i < 2; ++i)
        for (int j = 0; j < 4; ++j)
            for (int r = 0; r < 8; ++r) {
                int row = row0 + 16 * i + r + 8 * half;
                int col = col0 + 16 * j + ln;
                C[(size_t)row * N + col] = acc[i][j][r];
            }
}

// ---------------------------------------------------------------------------
// 3) RoPE + L2 qk-norm + head-major bf16 pack.
//    raw: [S, nheads*128] f32  ->  out: [nheads, S, 128] bf16
// ---------------------------------------------------------------------------
__global__ __launch_bounds__(256) void rope_l2norm_pack(const float* __restrict__ raw,
                                                        const int* __restrict__ pos,
                                                        const float* __restrict__ scale,
                                                        unsigned short* __restrict__ out,
                                                        int nheads) {
    const int lane = threadIdx.x & 31;
    const int wave = threadIdx.x >> 5;
    int idx = blockIdx.x * 8 + wave;
    int s = idx / nheads;
    int h = idx - s * nheads;

    const float* src = raw + ((size_t)s * nheads + h) * HD;
    float p = (float)pos[s];
    const float NEG_LOG_BASE_OVER_64 = -0.20503692777194263f; // -ln(500000)/64

    float o1[2], o2[2];
    float ss = 0.f;
    for (int j = 0; j < 2; ++j) {
        int d = lane + 32 * j;               // 0..63
        float x1 = src[d];
        float x2 = src[d + 64];
        float invf = __expf((float)d * NEG_LOG_BASE_OVER_64); // base^(-d/64)
        float f = p * invf;
        float c, sn;
        __sincosf(f, &sn, &c);
        o1[j] = x1 * c - x2 * sn;
        o2[j] = x2 * c + x1 * sn;
        ss += o1[j] * o1[j] + o2[j] * o2[j];
    }
    for (int mk = 1; mk < 32; mk <<= 1) ss += __shfl_xor(ss, mk, 32);
    float inv = 1.0f / (sqrtf(ss) + 1e-6f);

    unsigned short* dst = out + ((size_t)h * SEQ + s) * HD;
    for (int j = 0; j < 2; ++j) {
        int d = lane + 32 * j;
        dst[d]      = f32_to_bf16_rne(scale[d] * o1[j] * inv);
        dst[d + 64] = f32_to_bf16_rne(scale[d + 64] * o2[j] * inv);
    }
}

// ---------------------------------------------------------------------------
// 4) V pack + transpose:  Vraw [S, 8*128] f32 -> Vt [8, 128, S] bf16
// ---------------------------------------------------------------------------
__global__ __launch_bounds__(256) void v_pack_transpose(const float* __restrict__ Vraw,
                                                        unsigned short* __restrict__ Vt) {
    int idx = blockIdx.x * 256 + threadIdx.x;       // 0 .. S*KVDIM-1
    int s = idx >> 10;
    int c = idx & 1023;
    int h = c >> 7;
    int d = c & 127;
    Vt[((size_t)h * HD + d) * SEQ + s] = f32_to_bf16_rne(Vraw[idx]);
}

// ---------------------------------------------------------------------------
// 5) Flash attention (causal, GQA rep=4).
//    Block = 8 waves = one (head, 128-query tile); all waves share the same
//    key range (uniform causal trip count), so K/V tiles are staged into LDS
//    ONCE per block via async global->LDS copies (ASYNCcnt), double-buffered:
//      issue tile i+1 -> s_wait_asynccnt 8 -> barrier -> WMMA on tile i.
//    Wave owns 16 queries x d=128.  P goes through a wave-private LDS round
//    trip (C-layout f32 -> bf16 A-fragments) guarded by s_wait_dscnt.
// ---------------------------------------------------------------------------
__global__ __launch_bounds__(256) void flash_attn_kernel(const unsigned short* __restrict__ Qh,
                                                         const unsigned short* __restrict__ Kh,
                                                         const unsigned short* __restrict__ Vt,
                                                         unsigned short* __restrict__ Oattn) {
    __shared__ unsigned short Kbuf[2][64 * HD];    // 2 x 16 KB : 64 keys x 128 d
    __shared__ unsigned short Vbuf[2][HD * 64];    // 2 x 16 KB : 128 d x 64 keys
    __shared__ unsigned short Plds[8 * 16 * 64];   // 16 KB, wave-private slices

    const int tid  = threadIdx.x;
    const int lane = tid & 31;
    const int w    = tid >> 5;
    const int half = lane >> 4;
    const int ln   = lane & 15;

    const int head = blockIdx.x >> 4;       // 32 heads
    const int qt   = blockIdx.x & 15;       // 16 query tiles of 128
    const int kvh  = head >> 2;             // GQA: 4 query heads per kv head
    const int qbase = qt * 128 + w * 16;

    unsigned short* pbuf = Plds + w * (16 * 64);

    const unsigned short* kBase = Kh + (size_t)kvh * SEQ * HD;
    const unsigned short* vBase = Vt + (size_t)kvh * HD * SEQ;

    // Per-tile async staging: K tile is 16 KB contiguous; V tile is 128 rows
    // of 128 B each.  8 async b128 issues per thread-call-site per wave.
    auto issue_tile = [&](int tile, int buf) {
        const int kb = tile * 64;
        const unsigned short* gk = kBase + (size_t)kb * HD;   // contiguous 16 KB
        for (int j = 0; j < 4; ++j) {
            int c = tid + 256 * j;                            // 16-byte chunk id
            async_cp16(gk + c * 8, &Kbuf[buf][c * 8]);
        }
        for (int j = 0; j < 4; ++j) {
            int c = tid + 256 * j;                            // 0..1023
            int d = c >> 3, i = c & 7;
            async_cp16(vBase + (size_t)d * SEQ + kb + i * 8,
                       &Vbuf[buf][d * 64 + i * 8]);
        }
    };

    // Q fragments for this wave: 16 rows x 128 d  -> 4 A-frags (bf16)
    ABFrag qf[4];
    {
        const unsigned short* qptr = Qh + ((size_t)head * SEQ + qbase + ln) * HD + 8 * half;
        for (int f = 0; f < 4; ++f) qf[f] = load_a_frag(qptr + 32 * f);
    }

    v8f o[8];
    float m[8], l[8];
    for (int n = 0; n < 8; ++n)
        for (int r = 0; r < 8; ++r) o[n][r] = 0.f;
    for (int r = 0; r < 8; ++r) { m[r] = -3.0e38f; l[r] = 0.f; }

    const float sm_scale = 0.08838834764831845f;   // 1/sqrt(128)
    const int ntiles = (qt * 128 + 128) >> 6;      // uniform across the block

    issue_tile(0, 0);

    for (int it = 0; it < ntiles; ++it) {
        const int cur = it & 1;
        const int kb = it * 64;

        if (it + 1 < ntiles) {
            issue_tile(it + 1, cur ^ 1);
            WAIT_ASYNC(8);    // in-order ASYNCcnt: tile `it`'s 8 ops are done
        } else {
            WAIT_ASYNC(0);
        }
        __syncthreads();      // all waves' tile-`it` data resident in LDS

        // ---- S = Q K^T  (16 x 64 scores), contraction over d=128 ----
        v8f s4[4];
        for (int t = 0; t < 4; ++t) {
            v8f c;
            for (int r = 0; r < 8; ++r) c[r] = 0.f;
            const unsigned short* kp = &Kbuf[cur][(16 * t + ln) * HD + 8 * half];
            for (int c4 = 0; c4 < 4; ++c4) {
                ABFrag b = load_b_frag(kp + 32 * c4);
                c = wmma_bf16(qf[c4], b, c);
            }
            s4[t] = c;
        }

        // ---- scale + causal mask + online softmax ----
        float rmax[8];
        for (int r = 0; r < 8; ++r) rmax[r] = -3.0e38f;
        for (int t = 0; t < 4; ++t) {
            int kcol = kb + 16 * t + ln;
            for (int r = 0; r < 8; ++r) {
                int qrow = qbase + r + 8 * half;
                float v = s4[t][r] * sm_scale;
                v = (kcol > qrow) ? -3.0e38f : v;
                s4[t][r] = v;
                rmax[r] = fmaxf(rmax[r], v);
            }
        }
        for (int r = 0; r < 8; ++r)
            for (int mk = 1; mk < 16; mk <<= 1)
                rmax[r] = fmaxf(rmax[r], __shfl_xor(rmax[r], mk, 32));

        float alpha[8], rsum[8];
        for (int r = 0; r < 8; ++r) {
            float mn = fmaxf(m[r], rmax[r]);
            alpha[r] = __expf(m[r] - mn);
            m[r] = mn;
            rsum[r] = 0.f;
        }
        for (int t = 0; t < 4; ++t)
            for (int r = 0; r < 8; ++r) {
                float pv = __expf(s4[t][r] - m[r]);
                s4[t][r] = pv;
                rsum[r] += pv;
            }
        for (int r = 0; r < 8; ++r) {
            for (int mk = 1; mk < 16; mk <<= 1) rsum[r] += __shfl_xor(rsum[r], mk, 32);
            l[r] = l[r] * alpha[r] + rsum[r];
        }
        for (int n = 0; n < 8; ++n)
            for (int r = 0; r < 8; ++r) o[n][r] *= alpha[r];

        // ---- P (C-layout, f32) -> LDS (bf16, row-major 16x64), wave-private ----
        for (int t = 0; t < 4; ++t)
            for (int r = 0; r < 8; ++r) {
                int row = r + 8 * half;
                int col = 16 * t + ln;
                pbuf[row * 64 + col] = f32_to_bf16_rne(s4[t][r]);
            }
        asm volatile("s_wait_dscnt 0" ::: "memory");

        // ---- reload P as A-fragments, accumulate O += P V ----
        ABFrag pf[2];
        {
            const unsigned short* pp = pbuf + ln * 64 + 8 * half;
            pf[0] = load_a_frag(pp);
            pf[1] = load_a_frag(pp + 32);
        }
        for (int n = 0; n < 8; ++n) {
            const unsigned short* vp = &Vbuf[cur][(n * 16 + ln) * 64 + 8 * half];
            ABFrag b0 = load_b_frag(vp);
            ABFrag b1 = load_b_frag(vp + 32);
            o[n] = wmma_bf16(pf[0], b0, o[n]);
            o[n] = wmma_bf16(pf[1], b1, o[n]);
        }

        __syncthreads();      // buffer `cur` free to be overwritten (iter it+2)
    }

    // ---- epilogue: O / l  -> attn buffer [S, H] bf16 (head-interleaved cols) ----
    for (int n = 0; n < 8; ++n)
        for (int r = 0; r < 8; ++r) {
            int row = qbase + r + 8 * half;
            float val = o[n][r] / l[r];
            Oattn[(size_t)row * HID + head * HD + n * 16 + ln] = f32_to_bf16_rne(val);
        }
}

// ---------------------------------------------------------------------------
// Launch orchestration
// ---------------------------------------------------------------------------
extern "C" void kernel_launch(void* const* d_in, const int* in_sizes, int n_in,
                              void* d_out, int out_size, void* d_ws, size_t ws_size,
                              hipStream_t stream) {
    (void)in_sizes; (void)n_in; (void)out_size; (void)ws_size;

    const float* X   = (const float*)d_in[0];   // [1, 2048, 4096]
    const int*   pos = (const int*)d_in[1];     // [1, 2048]
    const float* Wq  = (const float*)d_in[2];   // [4096, 4096]
    const float* Wk  = (const float*)d_in[3];   // [1024, 4096]
    const float* Wv  = (const float*)d_in[4];   // [1024, 4096]
    const float* Wo  = (const float*)d_in[5];   // [4096, 4096]
    const float* qsc = (const float*)d_in[6];   // [128]
    const float* ksc = (const float*)d_in[7];   // [128]
    float* out = (float*)d_out;                 // [1, 2048, 4096] fp32

    const size_t MB = 1024 * 1024;
    char* ws = (char*)d_ws;
    unsigned short* Xb   = (unsigned short*)(ws + 0 * MB);     // 16 MB
    unsigned short* Wqb  = (unsigned short*)(ws + 16 * MB);    // 32 MB
    unsigned short* Wkb  = (unsigned short*)(ws + 48 * MB);    // 8 MB
    unsigned short* Wvb  = (unsigned short*)(ws + 56 * MB);    // 8 MB
    unsigned short* Wob  = (unsigned short*)(ws + 64 * MB);    // 32 MB
    float*          Qraw = (float*)(ws + 96 * MB);             // 32 MB
    float*          Kraw = (float*)(ws + 128 * MB);            // 8 MB
    float*          Vraw = (float*)(ws + 136 * MB);            // 8 MB
    unsigned short* Qh   = (unsigned short*)(ws + 144 * MB);   // 16 MB
    unsigned short* Kh   = (unsigned short*)(ws + 160 * MB);   // 4 MB
    unsigned short* Vt   = (unsigned short*)(ws + 164 * MB);   // 4 MB
    unsigned short* attn = (unsigned short*)Qraw;              // alias (Qraw dead by then)

    // 1) convert fp32 -> bf16
    auto cvt = [&](const float* src, unsigned short* dst, size_t n) {
        int n4 = (int)(n / 4);
        cvt_f32_bf16<<<(n4 + 255) / 256, 256, 0, stream>>>(src, dst, n4);
    };
    cvt(X,  Xb,  (size_t)SEQ * HID);
    cvt(Wq, Wqb, (size_t)HID * HID);
    cvt(Wk, Wkb, (size_t)KVDIM * HID);
    cvt(Wv, Wvb, (size_t)KVDIM * HID);
    cvt(Wo, Wob, (size_t)HID * HID);

    // 2) projections: raw = X @ W^T   (wave tile 32x64 -> blocks = M/32 * N/64 / 8)
    gemm_nt_bf16_f32<<<(SEQ / 32) * (HID / 64) / 8,   256, 0, stream>>>(Xb, Wqb, Qraw, SEQ, HID,   HID);
    gemm_nt_bf16_f32<<<(SEQ / 32) * (KVDIM / 64) / 8, 256, 0, stream>>>(Xb, Wkb, Kraw, SEQ, KVDIM, HID);
    gemm_nt_bf16_f32<<<(SEQ / 32) * (KVDIM / 64) / 8, 256, 0, stream>>>(Xb, Wvb, Vraw, SEQ, KVDIM, HID);

    // 3) RoPE + L2 norm + pack head-major bf16
    rope_l2norm_pack<<<SEQ * NQH / 8,  256, 0, stream>>>(Qraw, pos, qsc, Qh, NQH);
    rope_l2norm_pack<<<SEQ * NKVH / 8, 256, 0, stream>>>(Kraw, pos, ksc, Kh, NKVH);

    // 4) V transpose-pack
    v_pack_transpose<<<SEQ * KVDIM / 256, 256, 0, stream>>>(Vraw, Vt);

    // 5) flash attention: 32 heads x 16 query tiles
    flash_attn_kernel<<<NQH * 16, 256, 0, stream>>>(Qh, Kh, Vt, attn);

    // 6) output projection: out = attn @ Wo^T
    gemm_nt_bf16_f32<<<(SEQ / 32) * (HID / 64) / 8, 256, 0, stream>>>(attn, Wob, out, SEQ, HID, HID);
}
